// MultiHeadAttention_63170378989990
// MI455X (gfx1250) — compile-verified, matrix operand order checked by reading, not run on hardware
//
#include <hip/hip_runtime.h>
#include <math.h>

// MultiHeadAttention: B=2, S=2048, E=1024, H=16, D=64. bf16 WMMA everywhere.

#define B_ 2
#define S_ 2048
#define E_ 1024
#define H_ 16
#define D_ 64
#define BS (B_ * S_)          // 4096 rows
#define INV_SCALE 0.125f      // 1/sqrt(64)

typedef __attribute__((ext_vector_type(16))) __bf16 v16bf;
typedef __attribute__((ext_vector_type(8)))  float  v8f;

union Frag16 {              // one 16-bit A/B WMMA fragment (8 VGPRs)
    v16bf v;
    uint4 q[2];
    unsigned short h[16];
};
union FragC {               // f32 C/D fragment (8 VGPRs)
    v8f   v;
    float f[8];
};

__device__ inline unsigned short f32_to_bf16(float x) {
    unsigned int u = __float_as_uint(x);
    u += 0x7FFFu + ((u >> 16) & 1u);      // round-to-nearest-even
    return (unsigned short)(u >> 16);
}

__device__ inline v8f wmma_bf16(v16bf a, v16bf b, v8f c) {
    return __builtin_amdgcn_wmma_f32_16x16x32_bf16(
        /*neg_a=*/false, a, /*neg_b=*/false, b,
        /*c_mod=*/(short)0, c, /*reuse_a=*/false, /*reuse_b=*/false);
}

// Load a 16x32 bf16 fragment from a row-major [rows x ld] bf16 matrix.
// A-operand: row0 = M tile base.  B-operand: matrix must hold B^T (n-major),
// row0 = N tile base.  Two contiguous 16B chunks per lane (global_load_b128).
__device__ inline v16bf load_frag_rm(const unsigned short* base, int row0,
                                     int ld, int k0) {
    int lane = threadIdx.x & 31;
    int r    = row0 + (lane & 15);
    int kh   = (lane >> 4) << 3;          // 0 or 8
    const unsigned short* p = base + (size_t)r * ld + k0 + kh;
    Frag16 f;
    f.q[0] = *(const uint4*)(p);          // k = k0+kh .. +7
    f.q[1] = *(const uint4*)(p + 16);     // k = k0+16+kh .. +7
    return f.v;
}

// ---------------------------------------------------------------- converts
__global__ void cvt_f32_bf16(const float* __restrict__ src,
                             unsigned short* __restrict__ dst, int n) {
    int i = blockIdx.x * 256 + threadIdx.x;
    if (i < n) dst[i] = f32_to_bf16(src[i]);
}

// wt[n*1024 + k] = bf16(w[k*1024 + n])   (store W^T so B-frags are contiguous)
__global__ void transpose_w_bf16(const float* __restrict__ w,
                                 unsigned short* __restrict__ wt) {
    int i = blockIdx.x * 256 + threadIdx.x;   // over 1M elements
    int n = i >> 10, k = i & 1023;
    wt[i] = f32_to_bf16(w[k * E_ + n]);
}

// ------------------------------------------------------- projection GEMMs
// C[m,n] = sum_k A[m,k] * W[k,n], A bf16 [BS x 1024] row-major, BT = W^T bf16.
// mode 0: store bf16 into [B,H,S,D]; mode 1: store bf16 into [B,H,D,S].
__global__ __launch_bounds__(128)
void gemm_proj(const unsigned short* __restrict__ A,
               const unsigned short* __restrict__ BT,
               unsigned short* __restrict__ out, int mode) {
    int lane = threadIdx.x & 31;
    int w    = threadIdx.x >> 5;
    int row0 = blockIdx.x * 64 + (w >> 1) * 32;   // M tile (BS/64 blocks)
    int col0 = blockIdx.y * 64 + (w & 1) * 32;    // N tile (E/64 blocks)

    FragC c[2][2];
    for (int i = 0; i < 2; ++i)
        for (int j = 0; j < 2; ++j)
            for (int r = 0; r < 8; ++r) c[i][j].f[r] = 0.0f;

    for (int k0 = 0; k0 < E_; k0 += 32) {
        v16bf a0 = load_frag_rm(A,  row0,      E_, k0);
        v16bf a1 = load_frag_rm(A,  row0 + 16, E_, k0);
        v16bf b0 = load_frag_rm(BT, col0,      E_, k0);
        v16bf b1 = load_frag_rm(BT, col0 + 16, E_, k0);
        c[0][0].v = wmma_bf16(a0, b0, c[0][0].v);
        c[0][1].v = wmma_bf16(a0, b1, c[0][1].v);
        c[1][0].v = wmma_bf16(a1, b0, c[1][0].v);
        c[1][1].v = wmma_bf16(a1, b1, c[1][1].v);
    }

    int hi = (lane >> 4) << 3;                    // +8 rows for high half
    for (int i = 0; i < 2; ++i)
        for (int j = 0; j < 2; ++j)
            for (int r = 0; r < 8; ++r) {
                int m = row0 + i * 16 + r + hi;
                int n = col0 + j * 16 + (lane & 15);
                unsigned short hv = f32_to_bf16(c[i][j].f[r]);
                int b = m >> 11, s = m & 2047;    // m = b*2048 + s
                int h = n >> 6,  d = n & 63;      // n = h*64 + d
                size_t addr;
                if (mode == 0)                    // [B,H,S,D]
                    addr = (((size_t)(b * H_ + h) * S_) + s) * D_ + d;
                else                              // [B,H,D,S]
                    addr = (((size_t)(b * H_ + h) * D_) + d) * S_ + s;
                out[addr] = hv;
            }
}

// ------------------------------------------------------------- attention
// 1 block = (b*H+h, 32-query tile).  LDS: 32 x 2048 f32 score rows (256 KB).
__global__ __launch_bounds__(128)
void attention_kernel(const unsigned short* __restrict__ Qb,   // [B,H,S,D]
                      const unsigned short* __restrict__ Kb,   // [B,H,S,D]
                      const unsigned short* __restrict__ Vt,   // [B,H,D,S]
                      unsigned short* __restrict__ Obuf) {     // [BS,E]
    extern __shared__ char smem_raw[];
    float* sc  = (float*)smem_raw;            // [32][2048]
    float* red = sc + 32 * S_;                // [32][4]
    float* inv = red + 128;                   // [32]

    int lane  = threadIdx.x & 31;
    int w     = threadIdx.x >> 5;
    int bh    = blockIdx.x >> 6;              // b*H + h
    int qbase = (blockIdx.x & 63) * 32;

    const unsigned short* Qp = Qb + (size_t)bh * S_ * D_;
    const unsigned short* Kp = Kb + (size_t)bh * S_ * D_;
    const unsigned short* Vp = Vt + (size_t)bh * D_ * S_;

    // ---- pass 1: scores = mask(Q K^T / sqrt(D)); waves split the 2048 keys
    v16bf aq[2][2];                           // [qgroup][d-half]
    for (int g = 0; g < 2; ++g)
        for (int h = 0; h < 2; ++h)
            aq[g][h] = load_frag_rm(Qp, qbase + g * 16, D_, h * 32);

    int hi = (lane >> 4) << 3;
    for (int kt = w * 512; kt < w * 512 + 512; kt += 16) {
        v16bf b0 = load_frag_rm(Kp, kt, D_, 0);    // d 0..31  (K^T frag)
        v16bf b1 = load_frag_rm(Kp, kt, D_, 32);   // d 32..63
        for (int g = 0; g < 2; ++g) {
            FragC cs;
            for (int r = 0; r < 8; ++r) cs.f[r] = 0.0f;
            cs.v = wmma_bf16(aq[g][0], b0, cs.v);
            cs.v = wmma_bf16(aq[g][1], b1, cs.v);
            for (int r = 0; r < 8; ++r) {
                int row = g * 16 + r + hi;         // 0..31 within tile
                int col = kt + (lane & 15);        // key index
                float v = cs.f[r] * INV_SCALE;
                if (col > qbase + row) v = -1e9f;  // causal mask
                sc[row * S_ + col] = v;
            }
        }
    }
    __syncthreads();

    // ---- pass 2: row softmax, store p as bf16 in-place (low half of slot)
    {
        int row = threadIdx.x >> 2;
        int seg = threadIdx.x & 3;
        float* rp = sc + row * S_ + seg * 512;
        float m = -3.4e38f;
        for (int k = 0; k < 512; ++k) m = fmaxf(m, rp[k]);
        red[row * 4 + seg] = m;
        __syncthreads();
        m = fmaxf(fmaxf(red[row * 4 + 0], red[row * 4 + 1]),
                  fmaxf(red[row * 4 + 2], red[row * 4 + 3]));
        float s = 0.0f;
        for (int k = 0; k < 512; ++k) {
            float p = __expf(rp[k] - m);
            s += p;
            ((unsigned short*)&rp[k])[0] = f32_to_bf16(p);
        }
        red[row * 4 + seg] = s;                    // reuse after barrier
        __syncthreads();
        if (seg == 0) {
            float t = red[row * 4] + red[row * 4 + 1] +
                      red[row * 4 + 2] + red[row * 4 + 3];
            inv[row] = 1.0f / t;
        }
    }
    __syncthreads();

    // ---- pass 3: O = P V ; waves split D into 16-wide slabs
    int dbase = w * 16;
    FragC o[2];
    for (int g = 0; g < 2; ++g)
        for (int r = 0; r < 8; ++r) o[g].f[r] = 0.0f;

    for (int kb = 0; kb < S_; kb += 32) {
        v16bf bv = load_frag_rm(Vp, dbase, S_, kb);    // V^T rows = d
        for (int g = 0; g < 2; ++g) {
            // gather P A-fragment from bf16 halves stored in f32 LDS slots
            int r = g * 16 + (lane & 15);
            int kh = (lane >> 4) << 3;
            const float* rp = sc + r * S_ + kb;
            Frag16 f;
            #pragma unroll
            for (int j = 0; j < 8; ++j)
                f.h[j] = ((const unsigned short*)&rp[kh + j])[0];
            #pragma unroll
            for (int j = 0; j < 8; ++j)
                f.h[8 + j] = ((const unsigned short*)&rp[16 + kh + j])[0];
            o[g].v = wmma_bf16(f.v, bv, o[g].v);
        }
    }

    int b = bh >> 4, h = bh & 15;
    for (int g = 0; g < 2; ++g)
        for (int r = 0; r < 8; ++r) {
            int row = g * 16 + r + hi;
            int sg  = qbase + row;
            int col = h * D_ + dbase + (lane & 15);
            float v = o[g].f[r] * inv[row];
            Obuf[((size_t)(b * S_ + sg)) * E_ + col] = f32_to_bf16(v);
        }
}

// ------------------------------------------------- output GEMM (+ bias, f32)
__global__ __launch_bounds__(128)
void gemm_out(const unsigned short* __restrict__ A,     // Obuf bf16 [BS,E]
              const unsigned short* __restrict__ BT,    // Wo^T bf16
              const float* __restrict__ bias,
              float* __restrict__ out) {
    int lane = threadIdx.x & 31;
    int w    = threadIdx.x >> 5;
    int row0 = blockIdx.x * 64 + (w >> 1) * 32;
    int col0 = blockIdx.y * 64 + (w & 1) * 32;

    FragC c[2][2];
    for (int i = 0; i < 2; ++i)
        for (int j = 0; j < 2; ++j)
            for (int r = 0; r < 8; ++r) c[i][j].f[r] = 0.0f;

    for (int k0 = 0; k0 < E_; k0 += 32) {
        v16bf a0 = load_frag_rm(A,  row0,      E_, k0);
        v16bf a1 = load_frag_rm(A,  row0 + 16, E_, k0);
        v16bf b0 = load_frag_rm(BT, col0,      E_, k0);
        v16bf b1 = load_frag_rm(BT, col0 + 16, E_, k0);
        c[0][0].v = wmma_bf16(a0, b0, c[0][0].v);
        c[0][1].v = wmma_bf16(a0, b1, c[0][1].v);
        c[1][0].v = wmma_bf16(a1, b0, c[1][0].v);
        c[1][1].v = wmma_bf16(a1, b1, c[1][1].v);
    }

    int hi = (lane >> 4) << 3;
    for (int i = 0; i < 2; ++i)
        for (int j = 0; j < 2; ++j)
            for (int r = 0; r < 8; ++r) {
                int m = row0 + i * 16 + r + hi;
                int n = col0 + j * 16 + (lane & 15);
                out[(size_t)m * E_ + n] = c[i][j].f[r] + bias[n];
            }
}

// ---------------------------------------------------------------- launcher
extern "C" void kernel_launch(void* const* d_in, const int* in_sizes, int n_in,
                              void* d_out, int out_size, void* d_ws, size_t ws_size,
                              hipStream_t stream) {
    const float* q  = (const float*)d_in[0];
    const float* k  = (const float*)d_in[1];
    const float* v  = (const float*)d_in[2];
    // d_in[3] = mask (causal; reconstructed analytically)
    const float* Wq = (const float*)d_in[4];
    const float* Wk = (const float*)d_in[5];
    const float* Wv = (const float*)d_in[6];
    const float* Wo = (const float*)d_in[7];
    const float* bo = (const float*)d_in[8];
    float* out = (float*)d_out;
    (void)in_sizes; (void)n_in; (void)out_size; (void)ws_size;

    const size_t MSZ = (size_t)BS * E_;   // 4M elements
    const size_t WSZ = (size_t)E_ * E_;   // 1M elements
    unsigned short* Xq  = (unsigned short*)d_ws;
    unsigned short* Xk  = Xq  + MSZ;
    unsigned short* Xv  = Xk  + MSZ;
    unsigned short* WqT = Xv  + MSZ;
    unsigned short* WkT = WqT + WSZ;
    unsigned short* WvT = WkT + WSZ;
    unsigned short* WoT = WvT + WSZ;
    unsigned short* Qb  = WoT + WSZ;      // [B,H,S,D]
    unsigned short* Kb  = Qb  + MSZ;      // [B,H,S,D]
    unsigned short* Vt  = Kb  + MSZ;      // [B,H,D,S]
    unsigned short* Ob  = Vt  + MSZ;      // [BS,E]

    // phase 0: convert / transpose to bf16
    cvt_f32_bf16<<<(int)(MSZ / 256), 256, 0, stream>>>(q, Xq, (int)MSZ);
    cvt_f32_bf16<<<(int)(MSZ / 256), 256, 0, stream>>>(k, Xk, (int)MSZ);
    cvt_f32_bf16<<<(int)(MSZ / 256), 256, 0, stream>>>(v, Xv, (int)MSZ);
    transpose_w_bf16<<<(int)(WSZ / 256), 256, 0, stream>>>(Wq, WqT);
    transpose_w_bf16<<<(int)(WSZ / 256), 256, 0, stream>>>(Wk, WkT);
    transpose_w_bf16<<<(int)(WSZ / 256), 256, 0, stream>>>(Wv, WvT);
    transpose_w_bf16<<<(int)(WSZ / 256), 256, 0, stream>>>(Wo, WoT);

    // phase 1: Q/K/V projections (WMMA)
    dim3 ggrid(BS / 64, E_ / 64);
    gemm_proj<<<ggrid, 128, 0, stream>>>(Xq, WqT, Qb, 0);
    gemm_proj<<<ggrid, 128, 0, stream>>>(Xk, WkT, Kb, 0);
    gemm_proj<<<ggrid, 128, 0, stream>>>(Xv, WvT, Vt, 1);

    // phase 2: fused causal attention per (b,h,32-query tile)
    size_t smem = (size_t)(32 * S_ + 128 + 32) * sizeof(float);  // ~257 KB
    attention_kernel<<<B_ * H_ * (S_ / 32), 128, smem, stream>>>(Qb, Kb, Vt, Ob);

    // phase 3: output projection + bias (WMMA, f32 out)
    gemm_out<<<ggrid, 128, 0, stream>>>(Ob, WoT, bo, out);
}
